// DirectedSpatialConv_11235634446652
// MI455X (gfx1250) — matrix operand order repaired and checked by reading, not compiled, not run on hardware
//
#include <hip/hip_runtime.h>

// ---------------------------------------------------------------------------
// DirectedSpatialConv for MI455X (gfx1250), bf16x3-split WMMA implementation.
// out[b,o,n,t] = sum_k sum_c W[k,o,c] * ( (Abar_k @ x[b,c])[n,t]
//                                        + ((Pe_k+Ce_k) @ xo[b,c])[n,t] ) + bias
//
// Aggregation GEMM: A-operand = x tile (M=t, K=j/e), B-operand = operator
// (K=j/e, N=n). Projection GEMM: A-operand = W (M=o, K=c), B = agg (K=c, N=q).
// All products use bf16 hi/lo x3 (hi*hi + hi*lo + lo*hi) for ~fp32 accuracy.
// ---------------------------------------------------------------------------

typedef __attribute__((ext_vector_type(16))) __bf16 v16bf;
typedef __attribute__((ext_vector_type(8)))  float  v8f;
typedef __attribute__((ext_vector_type(4)))  float  v4f;
typedef __attribute__((ext_vector_type(4)))  unsigned int v4u;

union B32x8 {           // 32 bytes = one WMMA 16-bf16 operand fragment
  v4u   q[2];
  v16bf v;
};

// Split f32 into bf16 hi + bf16 lo (truncation; combined ~16 mantissa bits)
__device__ __forceinline__ void split2(float v, unsigned short& h, unsigned short& l) {
  unsigned u = __float_as_uint(v);
  h = (unsigned short)(u >> 16);
  float hv = __uint_as_float(u & 0xFFFF0000u);
  l = (unsigned short)(__float_as_uint(v - hv) >> 16);
}

// acc += (ah+al) * (bh+bl), dropping the negligible lo*lo term (bf16x3 trick)
__device__ __forceinline__ v8f wmma3(v16bf ah, v16bf al, v16bf bh, v16bf bl, v8f c) {
  c = __builtin_amdgcn_wmma_f32_16x16x32_bf16(false, ah, false, bh, (short)0, c, false, false);
  c = __builtin_amdgcn_wmma_f32_16x16x32_bf16(false, ah, false, bl, (short)0, c, false, false);
  c = __builtin_amdgcn_wmma_f32_16x16x32_bf16(false, al, false, bh, (short)0, c, false, false);
  return c;
}

// ------------------------- prep kernel 1: dinv + bias sum -------------------
__global__ void dsc_prep1(const float* __restrict__ Av, const float* __restrict__ bias,
                          float* __restrict__ dinv, float* __restrict__ bsum) {
  const int tid = threadIdx.x;          // 512 threads = 4*128 rows
  const int k = tid >> 7, n = tid & 127;
  const float* row = Av + ((size_t)(k * 128 + n)) * 128;
  float s = 0.f;
  for (int j = 0; j < 128; ++j) s += row[j];
  dinv[tid] = rsqrtf(s + 1e-3f);
  if (tid < 64) {
    float b = 0.f;
    for (int kk = 0; kk < 4; ++kk) b += bias[kk * 64 + tid];
    bsum[tid] = b;
  }
}

// ---- prep kernel 2: pack operators into WMMA operand fragments (bf16 hi/lo)
// B-operand layout (K=32 x N=16 bf16): lane l -> col N=l&15; elem i -> row
//   K=(l>>4)*16+i.   (Abar and Pe+Ce fragments)
// A-operand layout (M=16 x K=32 bf16): lane l -> row M=l&15; elem i -> col
//   K=(l>>4)*16+i.  (W fragments)
// Fragment storage: [frag][lane][0..15]=hi, [16..31]=lo (64B per lane).
__global__ void dsc_prep2(const float* __restrict__ Av, const float* __restrict__ Pe,
                          const float* __restrict__ Ce, const float* __restrict__ W,
                          const float* __restrict__ dinv,
                          unsigned short* __restrict__ aF,
                          unsigned short* __restrict__ pF,
                          unsigned short* __restrict__ wF) {
  const int f = blockIdx.x;             // 0..415
  const int lane = threadIdx.x;         // 0..31
  const int m = lane & 15;

  if (f < 128) {                        // Abar B-frags: [k][ntg8][jchunk4]
    const int k = f >> 5, ntg = (f >> 2) & 7, jc = f & 3;
    const int n = ntg * 16 + m;
    unsigned short* dst = aF + (((size_t)f * 32 + lane) << 5);
    const float dn = dinv[k * 128 + n];
    #pragma unroll
    for (int i = 0; i < 16; ++i) {
      const int j = jc * 32 + ((lane >> 4) << 4) + i;
      const float v = dn * Av[((size_t)(k * 128 + n)) * 128 + j] * dinv[k * 128 + j];
      unsigned short h, l;
      split2(v, h, l);
      dst[i] = h; dst[16 + i] = l;
    }
  } else if (f < 384) {                 // (Pe+Ce) B-frags: [k][ntg8][echunk8]
    const int g = f - 128;
    const int k = g >> 6, ntg = (g >> 3) & 7, ec = g & 7;
    const int n = ntg * 16 + m;
    unsigned short* dst = pF + (((size_t)g * 32 + lane) << 5);
    #pragma unroll
    for (int i = 0; i < 16; ++i) {
      const int e = ec * 32 + ((lane >> 4) << 4) + i;
      const size_t idx = ((size_t)(k * 128 + n)) * 256 + e;
      const float v = Pe[idx] + Ce[idx];
      unsigned short h, l;
      split2(v, h, l);
      dst[i] = h; dst[16 + i] = l;
    }
  } else {                              // W A-frags: [k][otile4][cchunk2]
    const int g = f - 384;
    const int k = g >> 3, ot = (g >> 1) & 3, cc = g & 1;
    const int o = ot * 16 + m;
    const int kbase = (lane >> 4) << 4;
    unsigned short* dst = wF + (((size_t)g * 32 + lane) << 5);
    #pragma unroll
    for (int i = 0; i < 16; ++i) {
      const int c = cc * 32 + kbase + i;
      const float v = W[((size_t)(k * 64 + o)) * 64 + c];
      unsigned short h, l;
      split2(v, h, l);
      dst[i] = h; dst[16 + i] = l;
    }
  }
}

// ------------------------------- main kernel --------------------------------
// Grid: (T/16, N/32, B). Block 512 threads (16 waves). Each WG: 32 nodes,
// 16 t columns, all 64 out-channels, all 4 subsets.
__launch_bounds__(512)
__global__ void dsc_main(const float* __restrict__ x, const float* __restrict__ xo,
                         const unsigned short* __restrict__ aF,
                         const unsigned short* __restrict__ pF,
                         const unsigned short* __restrict__ wFr,
                         const float* __restrict__ bsum,
                         float* __restrict__ out) {
  // LDS plan (282,624 B total):
  //  [0, 262144)      aggT[k4][q512][seg4][16] ushort : q = tloc*32 + nloc,
  //                   seg = {hi c0-15, lo c0-15, hi c16-31, lo c16-31}
  //                   (rows 128B -> every ds_load_b128 16B aligned)
  //                   epilogue reuse: obuf[o64][n32][20] float
  //  [262144, 282624) fragLDS[8][32][40] ushort : per-chunk operator frags
  __shared__ __align__(16) char smem[282624];
  typedef unsigned short aggT_t[512][4][16];
  aggT_t* aggT = (aggT_t*)smem;
  typedef unsigned short frag_t[32][40];
  frag_t* fragLDS = (frag_t*)(smem + 262144);
  float* obuf = (float*)smem;

  const int tid  = threadIdx.x;
  const int wave = tid >> 5;
  const int lane = tid & 31;
  const int l15  = lane & 15;
  const int lhi  = lane >> 4;           // 0/1
  const int t0   = blockIdx.x << 4;
  const int b    = blockIdx.z;

  const int otile = wave & 3;           // projection: this wave's o-tile
  const int qtb   = (wave >> 2) << 3;   // this wave's 8 q-tiles

  v8f outacc[8] = {};

  for (int chalf = 0; chalf < 2; ++chalf) {
    // ------------- aggregation fill: 32 channels, 2 per wave ---------------
    for (int cc = 0; cc < 2; ++cc) {
      const int c = chalf * 32 + wave * 2 + cc;
      const float* xb  = x  + ((size_t)(b * 64 + c) << 17);   // *128*1024
      const float* xob = xo + ((size_t)(b * 64 + c) << 18);   // *256*1024
      v8f agg[8] = {};                  // [k][ntile2]

      for (int ch = 0; ch < 12; ++ch) {            // 4 j-chunks + 8 e-chunks
        const bool isJ = ch < 4;
        const int  cb  = isJ ? (ch << 5) : ((ch - 4) << 5);
        const float* xbase = isJ ? xb : xob;

        __syncthreads();                // previous chunk's frag reads done
        {  // cooperative stage of 8 operator frags (k4 x ntile2), 16KB
          const int fi = tid >> 6;                 // frag 0..7
          const int r  = tid & 63;
          const int fl = r >> 1, half = r & 1;     // lane, 32B half
          const int k  = fi >> 1;
          const int ntg = (blockIdx.y << 1) + (fi & 1);
          const unsigned short* src = isJ
              ? aF + ((size_t)(((k * 8 + ntg) * 4 + ch) * 32 + fl) << 5) + (half << 4)
              : pF + ((size_t)(((k * 8 + ntg) * 8 + (ch - 4)) * 32 + fl) << 5) + (half << 4);
          const v4u* s = (const v4u*)src;
          v4u* d = (v4u*)&fragLDS[fi][fl][half << 4];
          d[0] = s[0]; d[1] = s[1];
        }
        __syncthreads();

        if (ch != 3 && ch < 11)                    // hint next chunk (same tensor)
          __builtin_prefetch(&xbase[((size_t)(cb + 32 + (lhi << 3)) << 10) + t0 + l15], 0, 0);

        // x as A-operand: lane -> row t=t0+l15; elems -> strided j (coalesced
        // across lanes), split to bf16 hi/lo in registers. No LDS staging.
        B32x8 ah, al;
        {
          unsigned short* hp = (unsigned short*)&ah;
          unsigned short* lp = (unsigned short*)&al;
          #pragma unroll
          for (int i = 0; i < 16; ++i) {
            const int j = cb + (lhi << 4) + i;
            const float v = xbase[((size_t)j << 10) + t0 + l15];
            split2(v, hp[i], lp[i]);
          }
        }

        #pragma unroll
        for (int k = 0; k < 4; ++k) {
          #pragma unroll
          for (int n2 = 0; n2 < 2; ++n2) {
            const v4u* pb = (const v4u*)&fragLDS[k * 2 + n2][lane][0];
            B32x8 bh, bl;
            bh.q[0] = pb[0]; bh.q[1] = pb[1];
            bl.q[0] = pb[2]; bl.q[1] = pb[3];
            agg[k * 2 + n2] = wmma3(ah.v, al.v, bh.v, bl.v, agg[k * 2 + n2]);
          }
        }
      }

      // D tile: M=tloc=v+lhi*8, N=nloc(16 per ntile). Store pre-split bf16.
      const int cidx = wave * 2 + cc;              // 0..31
      const int shh = (cidx >> 4) << 1;            // hi segment
      const int ci  = cidx & 15;
      #pragma unroll
      for (int k = 0; k < 4; ++k) {
        #pragma unroll
        for (int n2 = 0; n2 < 2; ++n2) {
          const v8f a = agg[k * 2 + n2];
          #pragma unroll
          for (int v = 0; v < 8; ++v) {
            const int tloc = v + (lhi << 3);
            const int nloc = (n2 << 4) + l15;
            const int q = (tloc << 5) + nloc;      // t-major: conflict-free
            unsigned short h, l;
            split2(a[v], h, l);
            aggT[k][q][shh][ci]     = h;
            aggT[k][q][shh + 1][ci] = l;
          }
        }
      }
    }
    __syncthreads();

    // ---------- projection: out[o,q] += W[o,c] * agg[c,q], K=32 c ----------
    #pragma unroll
    for (int k = 0; k < 4; ++k) {
      const v4u* pw = (const v4u*)(wFr + ((size_t)((((k << 2) + otile) * 2 + chalf) * 32 + lane) << 5));
      B32x8 wh, wl;
      wh.q[0] = pw[0]; wh.q[1] = pw[1];
      wl.q[0] = pw[2]; wl.q[1] = pw[3];
      #pragma unroll
      for (int qq = 0; qq < 8; ++qq) {
        const int q = ((qtb + qq) << 4) + l15;
        const v4u* ph = (const v4u*)&aggT[k][q][lhi << 1][0];       // hi seg
        const v4u* pl = (const v4u*)&aggT[k][q][(lhi << 1) + 1][0]; // lo seg
        B32x8 bh, bl;
        bh.q[0] = ph[0]; bh.q[1] = ph[1];
        bl.q[0] = pl[0]; bl.q[1] = pl[1];
        outacc[qq] = wmma3(wh.v, wl.v, bh.v, bl.v, outacc[qq]);
      }
    }
    __syncthreads();                   // aggT refilled next chalf
  }

  // -------- epilogue: bounce through LDS so global stores coalesce --------
  #pragma unroll
  for (int qq = 0; qq < 8; ++qq) {
    const int qt = qtb + qq;
    const int tloc = qt >> 1;
    const int nloc = ((qt & 1) << 4) + l15;
    #pragma unroll
    for (int v = 0; v < 8; ++v) {
      const int o = (otile << 4) + (lhi << 3) + v;
      obuf[(o * 32 + nloc) * 20 + tloc] = outacc[qq][v];
    }
  }
  __syncthreads();
  #pragma unroll
  for (int p = 0; p < 4; ++p) {
    const int pid = (p << 9) + tid;    // 0..2047 = (o,n) pairs
    const int o = pid >> 5, n = pid & 31;
    const float bb = bsum[o];
    const float* r = &obuf[(o * 32 + n) * 20];
    v4f w0 = ((const v4f*)r)[0] + bb;
    v4f w1 = ((const v4f*)r)[1] + bb;
    v4f w2 = ((const v4f*)r)[2] + bb;
    v4f w3 = ((const v4f*)r)[3] + bb;
    float* g = out + ((((size_t)(b * 64 + o)) << 7) + (blockIdx.y << 5) + n) * 1024 + t0;
    ((v4f*)g)[0] = w0;
    ((v4f*)g)[1] = w1;
    ((v4f*)g)[2] = w2;
    ((v4f*)g)[3] = w3;
  }
}

// ----------------------------------------------------------------------------
extern "C" void kernel_launch(void* const* d_in, const int* in_sizes, int n_in,
                              void* d_out, int out_size, void* d_ws, size_t ws_size,
                              hipStream_t stream) {
  const float* x    = (const float*)d_in[0];   // (8,64,128,1024)
  const float* xo   = (const float*)d_in[1];   // (8,64,256,1024)
  const float* Av   = (const float*)d_in[2];   // (4,128,128)
  const float* Pe   = (const float*)d_in[3];   // (4,128,256)
  const float* Ce   = (const float*)d_in[4];   // (4,128,256)
  const float* W    = (const float*)d_in[5];   // (4,64,64)
  const float* bias = (const float*)d_in[6];   // (4,64)
  float* out = (float*)d_out;                  // (8,64,128,1024)

  // workspace layout (~855 KB)
  char* ws = (char*)d_ws;
  unsigned short* aF = (unsigned short*)(ws);                 // 262144 B
  unsigned short* pF = (unsigned short*)(ws + 262144);        // 524288 B
  unsigned short* wF = (unsigned short*)(ws + 786432);        //  65536 B
  float* bsum = (float*)(ws + 851968);                        //    256 B
  float* dinv = (float*)(ws + 852224);                        //   2048 B

  dsc_prep1<<<1, 512, 0, stream>>>(Av, bias, dinv, bsum);
  dsc_prep2<<<416, 32, 0, stream>>>(Av, Pe, Ce, W, dinv, aF, pF, wF);
  dsc_main<<<dim3(1024 / 16, 128 / 32, 8), 512, 0, stream>>>(x, xo, aF, pF, wF, bsum, out);
}
